// NeuralNetwork_81149112090731
// MI455X (gfx1250) — compile-verified, matrix operand order checked by reading, not run on hardware
//
#include <hip/hip_runtime.h>
#include <hip/hip_bf16.h>
#include <math.h>

typedef __attribute__((ext_vector_type(16))) _Float16 v16h;
typedef __attribute__((ext_vector_type(8)))  float    v8f;

#define BATCH      32768
#define C_OUT      256
#define FLATD      3072      // 256 * 12
#define NPOS       12
#define SMP        16        // samples per block in main kernel
#define KT1        96        // 3072 / 32 K-tiles for FC1
#define NT1        8         // 128 / 16 N-tiles for FC1

// ---------------------------------------------------------------------------
// Fragment swizzle helpers (CDNA5 ISA 7.12.2, 16-bit A/B matrices, wave32):
//  lane m (0..15)  holds K in {0..7, 16..23}  (element e: e<8 -> K=e, else 16+e-8)
//  lane m+16       holds K in {8..15, 24..31} (element e: e<8 -> K=8+e, else 24+e-8)
// ---------------------------------------------------------------------------
__device__ __forceinline__ int frag_k(int lane, int e) {
    int base = (lane & 16) ? 8 : 0;
    return (e < 8) ? (base + e) : (base + 16 + (e - 8));
}
// inverse: given K (0..31) -> which lane-half and element slot
__device__ __forceinline__ void frag_pos(int kl, int& laneHi, int& e) {
    laneHi = (kl >> 3) & 1;
    e = (kl & 7) + ((kl >> 4) & 1) * 8;
}

__device__ __forceinline__ v8f wmma_f16(v16h a, v16h b, v8f c) {
    return __builtin_amdgcn_wmma_f32_16x16x32_f16(false, a, false, b, (short)0, c,
                                                  false, false);
}

// raw LDS byte offset of a __shared__ object (what VDST of async-to-LDS wants)
__device__ __forceinline__ unsigned lds_off(const void* p) {
    return (unsigned)(size_t)(__attribute__((address_space(3))) const void*)p;
}

// async copy: 16 bytes per lane, global -> LDS, tracked by ASYNCcnt
__device__ __forceinline__ void async_b128(unsigned lds_byte_off, const void* gaddr) {
    asm volatile("global_load_async_to_lds_b128 %0, %1, off"
                 :: "v"(lds_byte_off), "v"((unsigned long long)(size_t)gaddr)
                 : "memory");
}
template<int N>
__device__ __forceinline__ void wait_async() {
    asm volatile("s_wait_asynccnt %0" :: "n"(N) : "memory");
}

// ===========================================================================
// Kernel 0: zero the stats accumulator (m[16] + S[256] floats)
// ===========================================================================
__global__ __launch_bounds__(512) void k_zero_stats(float* stat) {
    int t = threadIdx.x;
    if (t < 272) stat[t] = 0.0f;
}

// ===========================================================================
// Kernel 1: patch statistics.  thread t owns pair (k=t/16, j=t%16); block
// stages 64 samples in LDS.  Accumulates S[k][j] = sum p_k*p_j and m[k].
// ===========================================================================
__global__ __launch_bounds__(256) void k_stats(const float* __restrict__ x,
                                               float* __restrict__ stat) {
    __shared__ float xs[64 * 42];
    int t  = threadIdx.x;
    int b0 = blockIdx.x * 64;
    for (int i = t; i < 64 * 42; i += 256) xs[i] = x[b0 * 42 + i];
    __syncthreads();

    int k = t >> 4, j = t & 15;
    int kh = k >> 2, kw = k & 3, jh = j >> 2, jw = j & 3;
    float accS = 0.0f, accM = 0.0f;
    for (int s = 0; s < 64; ++s) {
        const float* xp = xs + s * 42;
        #pragma unroll
        for (int p = 0; p < NPOS; ++p) {
            int oh = p >> 2, ow = p & 3;
            float a = xp[(oh + kh) * 7 + (ow + kw)];
            float b = xp[(oh + jh) * 7 + (ow + jw)];
            accS = fmaf(a, b, accS);
            if (j == 0) accM += a;
        }
    }
    atomicAdd(stat + 16 + t, accS);
    if (j == 0) atomicAdd(stat + k, accM);
}

// ===========================================================================
// Kernel 2: fold BN into conv weights; pack w_eff into WMMA B-fragments.
// ===========================================================================
__global__ __launch_bounds__(256) void k_finalize(const float* __restrict__ stat,
                                                  const float* __restrict__ conv_w,
                                                  const float* __restrict__ conv_b,
                                                  const float* __restrict__ gamma,
                                                  const float* __restrict__ beta,
                                                  float* __restrict__ beff,
                                                  _Float16* __restrict__ convpack) {
    __shared__ float m_s[16];
    __shared__ float S_s[256];
    __shared__ float scale_s[256];
    int t = threadIdx.x;
    const float invN = 1.0f / ((float)BATCH * (float)NPOS);
    if (t < 16) m_s[t] = stat[t] * invN;
    S_s[t] = stat[16 + t] * invN;
    __syncthreads();

    {   // one channel per thread
        int c = t;
        float w[16];
        #pragma unroll
        for (int k = 0; k < 16; ++k) w[k] = conv_w[c * 16 + k];
        float wm = 0.0f;
        #pragma unroll
        for (int k = 0; k < 16; ++k) wm = fmaf(w[k], m_s[k], wm);
        float q = 0.0f;
        for (int k = 0; k < 16; ++k) {
            float acc = 0.0f;
            #pragma unroll
            for (int j = 0; j < 16; ++j) acc = fmaf(S_s[k * 16 + j], w[j], acc);
            q = fmaf(w[k], acc, q);
        }
        float cb   = conv_b[c];
        float mean = wm + cb;
        float ey2  = q + 2.0f * cb * wm + cb * cb;
        float var  = ey2 - mean * mean;
        float sc   = gamma[c] * rsqrtf(var + 1e-5f);
        scale_s[c] = sc;
        beff[c]    = beta[c] + sc * (cb - mean);
    }
    __syncthreads();

    // pack scaled conv weights: 16 tiles x 32 lanes x 16 elements
    for (int i = t; i < 16 * 512; i += 256) {
        int ct   = i >> 9;
        int lane = (i >> 4) & 31;
        int e    = i & 15;
        int c    = ct * 16 + (lane & 15);
        int k    = frag_k(lane, e);
        float v  = (k < 16) ? conv_w[c * 16 + k] * scale_s[c] : 0.0f;
        convpack[i] = (_Float16)v;
    }
}

// ===========================================================================
// Kernel 3: pack [pw1 | vw1]  (3072 x 128) into B fragments, f16.
// ===========================================================================
__global__ __launch_bounds__(256) void k_pack_w1(const float* __restrict__ pw1,
                                                 const float* __restrict__ vw1,
                                                 _Float16* __restrict__ w1pack) {
    int g = blockIdx.x * 256 + threadIdx.x;
    if (g >= FLATD * 128) return;
    int k = g >> 7, n = g & 127;
    float v = (n < 64) ? pw1[k * 64 + n] : vw1[k * 64 + (n - 64)];
    int kt = k >> 5, kl = k & 31;
    int laneHi, e;
    frag_pos(kl, laneHi, e);
    int nt = n >> 4, nl = n & 15;
    int lane = laneHi * 16 + nl;
    w1pack[(kt * NT1 + nt) * 512 + lane * 16 + e] = (_Float16)v;
}

// ===========================================================================
// Kernel 4: pack pw2 / vw2 (64 x 32 each) into B fragments.
// ===========================================================================
__global__ __launch_bounds__(256) void k_pack_w2(const float* __restrict__ pw2,
                                                 const float* __restrict__ vw2,
                                                 _Float16* __restrict__ w2pack) {
    int g = blockIdx.x * 256 + threadIdx.x;
    if (g >= 2 * 64 * 32) return;
    int head = g >> 11;
    int r = g & 2047;
    int k = r >> 5, n = r & 31;
    float v = head ? vw2[k * 32 + n] : pw2[k * 32 + n];
    int kt2 = k >> 5, kl = k & 31;
    int laneHi, e;
    frag_pos(kl, laneHi, e);
    int nt2 = n >> 4, nl = n & 15;
    int lane = laneHi * 16 + nl;
    w2pack[head * 2048 + (kt2 * 2 + nt2) * 512 + lane * 16 + e] = (_Float16)v;
}

// ===========================================================================
// Kernel 5: fused conv+BN+ReLU -> FC1 -> FC2 -> FC3 (+softmax/tanh)
// 16 samples per block, 8 waves.  FC1 weights streamed into LDS with
// global_load_async_to_lds_b128: wave-private 4-deep pipeline (3 tiles in
// flight), ASYNCcnt-gated; two accumulators break the WMMA dependency chain.
// ===========================================================================
__global__ __launch_bounds__(256) void k_main(const float* __restrict__ x,
                                              const float* __restrict__ pb1,
                                              const float* __restrict__ vb1,
                                              const float* __restrict__ pb2,
                                              const float* __restrict__ vb2,
                                              const float* __restrict__ pw3,
                                              const float* __restrict__ pb3,
                                              const float* __restrict__ vw3,
                                              const float* __restrict__ vb3,
                                              const _Float16* __restrict__ convpack,
                                              const float* __restrict__ beff,
                                              const _Float16* __restrict__ w1pack,
                                              const _Float16* __restrict__ w2pack,
                                              float* __restrict__ out) {
    __shared__ alignas(32) float    xs[SMP * 42];            // 2.6 KB
    __shared__ alignas(32) _Float16 hsf[KT1 * 512];          // 96 KB: h in A-frag layout
    __shared__ alignas(32) _Float16 wbuf[8][4][512];         // 32 KB: FC1 B 4-deep buffer
    __shared__ alignas(32) _Float16 f2[4 * 512];             // FC1 out, A-frag layout
    __shared__ alignas(32) float    sbuf[2 * SMP * 32];      // FC2 out

    const int t    = threadIdx.x;
    const int lane = t & 31;
    const int wave = t >> 5;
    const int b0   = blockIdx.x * SMP;

    // -------- stage 16 samples of x via async copy (168 x 16B = 2688B) ------
    if (t < 168)
        async_b128(lds_off(xs) + t * 16, (const char*)(x + (size_t)b0 * 42) + t * 16);
    wait_async<0>();
    __syncthreads();

    // -------- conv as WMMA: build A once per position, 2 channel tiles/wave --
    const int m     = lane & 15;          // sample row this lane owns
    const int sBase = (lane >> 4) << 3;   // +8 samples for hi half-wave in D
    {
        v16h bf0 = *(const v16h*)(convpack + wave * 512 + lane * 16);
        v16h bf1 = *(const v16h*)(convpack + (wave + 8) * 512 + lane * 16);
        const int c0 = wave * 16 + m;
        const int c1 = (wave + 8) * 16 + m;
        const float be0 = beff[c0];
        const float be1 = beff[c1];
        for (int p = 0; p < NPOS; ++p) {
            const int oh = p >> 2, ow = p & 3;
            v16h a;
            #pragma unroll
            for (int e = 0; e < 16; ++e) {
                int k = frag_k(lane, e);
                float v = 0.0f;
                if (k < 16) {
                    int kh = k >> 2, kw = k & 3;
                    v = xs[m * 42 + (oh + kh) * 7 + (ow + kw)];
                }
                a[e] = (_Float16)v;
            }
            v8f acc0 = {}, acc1 = {};
            acc0 = wmma_f16(a, bf0, acc0);
            acc1 = wmma_f16(a, bf1, acc1);
            // bias + ReLU + scatter into h fragment layout: col = c*12 + p
            #pragma unroll
            for (int half = 0; half < 2; ++half) {
                const int   c   = half ? c1 : c0;
                const float be  = half ? be1 : be0;
                const v8f   acc = half ? acc1 : acc0;
                const int col = c * NPOS + p;
                const int kt  = col >> 5, kl = col & 31;
                int laneHi, e2;
                frag_pos(kl, laneHi, e2);
                #pragma unroll
                for (int r = 0; r < 8; ++r) {
                    int sample = r + sBase;
                    float v = fmaxf(acc[r] + be, 0.0f);
                    hsf[kt * 512 + (laneHi * 16 + sample) * 16 + e2] = (_Float16)v;
                }
            }
        }
    }
    __syncthreads();

    // -------- FC1: (16 x 3072) @ (3072 x 128), wave = N-tile ----------------
    {
        const int nt = wave;
        _Float16* myw = &wbuf[wave][0][0];
        auto issue_tile = [&](int kt) {
            const _Float16* g = w1pack + (size_t)(kt * NT1 + nt) * 512;
            _Float16* d = myw + (kt & 3) * 512;
            #pragma unroll
            for (int i = 0; i < 2; ++i)
                async_b128(lds_off(d + i * 256 + lane * 8), g + i * 256 + lane * 8);
        };
        v8f acc0 = {}, acc1 = {};
        auto step = [&](int kt, v8f& acc) {
            v16h a = *(const v16h*)(hsf + kt * 512 + lane * 16);
            v16h b = *(const v16h*)(myw + (kt & 3) * 512 + lane * 16);
            acc = wmma_f16(a, b, acc);
        };
        issue_tile(0);
        issue_tile(1);
        issue_tile(2);
        // steady state: keep 3 tiles in flight; wait<=6 => current tile landed
        for (int kt = 0; kt < KT1 - 6; kt += 2) {
            issue_tile(kt + 3);
            wait_async<6>();
            step(kt, acc0);
            issue_tile(kt + 4);
            wait_async<6>();
            step(kt + 1, acc1);
        }
        // epilogue: kt = 90..95, drain the pipeline with exact counts
        issue_tile(93); wait_async<6>(); step(90, acc0);
        issue_tile(94); wait_async<6>(); step(91, acc1);
        issue_tile(95); wait_async<6>(); step(92, acc0);
        wait_async<4>();                 step(93, acc1);
        wait_async<2>();                 step(94, acc0);
        wait_async<0>();                 step(95, acc1);
        v8f acc = acc0 + acc1;

        const int gc   = nt * 16 + m;                 // 0..127
        const float bias = (gc < 64) ? pb1[gc] : vb1[gc - 64];
        const int head = gc >> 6;
        const int c2   = gc & 63;
        const int kt2  = c2 >> 5, kl = c2 & 31;
        int laneHi, e2;
        frag_pos(kl, laneHi, e2);
        #pragma unroll
        for (int r = 0; r < 8; ++r) {
            int sample = r + sBase;
            float v = fmaxf(acc[r] + bias, 0.0f);
            f2[head * 1024 + kt2 * 512 + (laneHi * 16 + sample) * 16 + e2] = (_Float16)v;
        }
    }
    __syncthreads();

    // -------- FC2: (16 x 64) @ (64 x 32) per head; waves 0..3 ---------------
    if (wave < 4) {
        const int head = wave >> 1;
        const int nt2  = wave & 1;
        v8f acc = {};
        #pragma unroll
        for (int kt2 = 0; kt2 < 2; ++kt2) {
            v16h a = *(const v16h*)(f2 + head * 1024 + kt2 * 512 + lane * 16);
            v16h b = *(const v16h*)(w2pack + head * 2048 + (kt2 * 2 + nt2) * 512 +
                                    lane * 16);
            acc = wmma_f16(a, b, acc);
        }
        const int c3 = nt2 * 16 + m;
        const float bias = head ? vb2[c3] : pb2[c3];
        #pragma unroll
        for (int r = 0; r < 8; ++r) {
            int sample = r + sBase;
            sbuf[head * (SMP * 32) + sample * 32 + c3] = fmaxf(acc[r] + bias, 0.0f);
        }
    }
    __syncthreads();

    // -------- FC3 + softmax (policy), tanh (value) --------------------------
    if (t < SMP) {
        const int sample = t;
        float logits[7];
        float mx = -1e30f;
        #pragma unroll
        for (int j = 0; j < 7; ++j) {
            float s = pb3[j];
            for (int k = 0; k < 32; ++k)
                s = fmaf(sbuf[sample * 32 + k], pw3[k * 7 + j], s);
            logits[j] = s;
            mx = fmaxf(mx, s);
        }
        float sum = 0.0f;
        #pragma unroll
        for (int j = 0; j < 7; ++j) {
            float e = __expf(logits[j] - mx);
            logits[j] = e;
            sum += e;
        }
        float inv = 1.0f / sum;
        #pragma unroll
        for (int j = 0; j < 7; ++j)
            out[(size_t)(b0 + sample) * 7 + j] = logits[j] * inv;
    } else if (t < 2 * SMP) {
        const int sample = t - SMP;
        float s = vb3[0];
        for (int k = 0; k < 32; ++k)
            s = fmaf(sbuf[SMP * 32 + sample * 32 + k], vw3[k], s);
        out[(size_t)BATCH * 7 + b0 + sample] = tanhf(s);
    }
}

// ===========================================================================
extern "C" void kernel_launch(void* const* d_in, const int* in_sizes, int n_in,
                              void* d_out, int out_size, void* d_ws, size_t ws_size,
                              hipStream_t stream) {
    const float* x       = (const float*)d_in[0];
    const float* conv_w  = (const float*)d_in[1];
    const float* conv_b  = (const float*)d_in[2];
    const float* gamma   = (const float*)d_in[3];
    const float* beta    = (const float*)d_in[4];
    const float* pw1     = (const float*)d_in[5];
    const float* pb1     = (const float*)d_in[6];
    const float* pw2     = (const float*)d_in[7];
    const float* pb2     = (const float*)d_in[8];
    const float* pw3     = (const float*)d_in[9];
    const float* pb3     = (const float*)d_in[10];
    const float* vw1     = (const float*)d_in[11];
    const float* vb1     = (const float*)d_in[12];
    const float* vw2     = (const float*)d_in[13];
    const float* vb2     = (const float*)d_in[14];
    const float* vw3     = (const float*)d_in[15];
    const float* vb3     = (const float*)d_in[16];
    float* out = (float*)d_out;

    char* ws = (char*)d_ws;
    float*    stat     = (float*)(ws + 0);            // 272 floats
    float*    beff     = (float*)(ws + 2048);         // 256 floats
    _Float16* convpack = (_Float16*)(ws + 4096);      // 16*512 halfs (16 KB)
    _Float16* w1pack   = (_Float16*)(ws + 20480);     // 96*8*512 halfs (768 KB)
    _Float16* w2pack   = (_Float16*)(ws + 20480 + 786432); // 2*4*512 halfs (8 KB)

    k_zero_stats<<<1, 512, 0, stream>>>(stat);
    k_stats<<<BATCH / 64, 256, 0, stream>>>(x, stat);
    k_finalize<<<1, 256, 0, stream>>>(stat, conv_w, conv_b, gamma, beta, beff, convpack);
    k_pack_w1<<<(FLATD * 128 + 255) / 256, 256, 0, stream>>>(pw1, vw1, w1pack);
    k_pack_w2<<<(2 * 64 * 32 + 255) / 256, 256, 0, stream>>>(pw2, vw2, w2pack);
    k_main<<<BATCH / SMP, 256, 0, stream>>>(x, pb1, vb1, pb2, vb2, pw3, pb3, vw3, vb3,
                                            convpack, beff, w1pack, w2pack, out);
}
